// CombinedVectorField_3710851744456
// MI455X (gfx1250) — matrix operand order repaired and checked by reading, not compiled, not run on hardware
//
#include <hip/hip_runtime.h>
#include <hip/hip_fp16.h>

typedef __attribute__((ext_vector_type(16))) _Float16 v16h;
typedef __attribute__((ext_vector_type(8)))  _Float16 v8h;
typedef __attribute__((ext_vector_type(8)))  float    v8f;

#define DIM_X 128
#define DIM_H 128
#define HID   512
#define BATCH 4096
#define ZK    256      // GEMM1 K: [x|h]; t-column folded into bias
#define ZSTR  264      // padded LDS row stride (halves) for z   (528B = 33*16)
#define ASTR  520      // padded LDS row stride (halves) for act (1040B = 65*16)

// ---- workspace layout (bytes) ----
#define WS_W1T   0                        // 512x256 f16: row n (hidden col), contiguous K
#define WS_W2T   (512*256*2)              // 128x512 f16: row n (out col), contiguous K
#define WS_B1E   (WS_W2T + 128*512*2)     // 512 f32: b1 + t*W1[256,:]
#define WS_S     (WS_B1E + 512*4)         // 512 f32: s[k] = sum_i W1[i,k]*W2[k,i]

__device__ __forceinline__ float tanh_fast(float x) {
  float e = __expf(-2.0f * fabsf(x));
  float r = (1.0f - e) / (1.0f + e);
  return copysignf(r, x);
}

__device__ __forceinline__ v16h combine8(v8h lo, v8h hi) {
  return __builtin_shufflevector(lo, hi, 0,1,2,3,4,5,6,7,8,9,10,11,12,13,14,15);
}

// -------- prep: fold t into bias, precompute divergence weights, f16-transpose weights --------
__global__ void prep_kernel(const float* __restrict__ W1, const float* __restrict__ b1,
                            const float* __restrict__ W2, const float* __restrict__ t,
                            _Float16* __restrict__ W1T, _Float16* __restrict__ W2T,
                            float* __restrict__ b1e, float* __restrict__ s) {
  const int n   = blockIdx.x;    // hidden unit 0..511
  const int tid = threadIdx.x;   // 128 threads
  // s[n] = sum_{i<128} W1[i,n] * W2[n,i]
  float p = W1[tid * HID + n] * W2[n * DIM_X + tid];
  for (int m = 1; m < 32; m <<= 1) p += __shfl_xor(p, m, 32);
  __shared__ float red[4];
  if ((tid & 31) == 0) red[tid >> 5] = p;
  __syncthreads();
  if (tid == 0) {
    s[n]   = red[0] + red[1] + red[2] + red[3];
    b1e[n] = b1[n] + t[0] * W1[256 * HID + n];
  }
  // W1T[n][k] = W1[k][n], k<256
  for (int k = tid; k < ZK; k += 128)
    W1T[n * ZK + k] = (_Float16)W1[k * HID + n];
  // W2T[n][k] = W2[k][n], n<128, k<512
  if (n < DIM_X)
    for (int k = tid; k < HID; k += 128)
      W2T[n * HID + k] = (_Float16)W2[k * DIM_X + n];
}

// -------- fused: GEMM1 (WMMA) -> tanh + divergence -> GEMM2 (WMMA) -> CFG blend --------
__global__ __launch_bounds__(256)
void fused_cfg_kernel(const float* __restrict__ state, const float* __restrict__ h,
                      const float* __restrict__ hnull, const float* __restrict__ gsp,
                      const float* __restrict__ b2,
                      const _Float16* __restrict__ W1T, const _Float16* __restrict__ W2T,
                      const float* __restrict__ b1e, const float* __restrict__ s,
                      float* __restrict__ out) {
  __shared__ __align__(16) _Float16 zl[2][16 * ZSTR];   // ctx0 = h, ctx1 = h_null
  __shared__ __align__(16) _Float16 al[2][16 * ASTR];   // tanh activations, f16
  __shared__ float sdiv[2][16];

  const int tid  = threadIdx.x;
  const int wave = tid >> 5;
  const int lane = tid & 31;
  const int lg   = lane >> 4;          // half-wave group (K/row-half select)
  const int ln   = lane & 15;
  const int r0   = blockIdx.x * 16;    // batch row tile

  if (tid < 32) sdiv[tid >> 4][tid & 15] = 0.0f;

  // stage z = [x | h] as f16 for both contexts
  for (int it = 0; it < 16; ++it) {
    int e = it * 256 + tid;
    int r = e >> 8, c = e & 255;
    if (c < DIM_X) {
      _Float16 v = (_Float16)state[(r0 + r) * (DIM_X + 1) + c];
      zl[0][r * ZSTR + c] = v;
      zl[1][r * ZSTR + c] = v;
    } else {
      zl[0][r * ZSTR + c] = (_Float16)h[(r0 + r) * DIM_H + (c - DIM_X)];
      zl[1][r * ZSTR + c] = (_Float16)hnull[(r0 + r) * DIM_H + (c - DIM_X)];
    }
  }
  __syncthreads();

  // ---- GEMM1: z(16x256) @ W1(256x512); wave owns 4 N-tiles of 16 ----
  float divacc[2][8];
  #pragma unroll
  for (int c = 0; c < 2; ++c)
    #pragma unroll
    for (int v = 0; v < 8; ++v) divacc[c][v] = 0.0f;

  for (int tt = 0; tt < 4; ++tt) {
    const int n0 = wave * 64 + tt * 16;
    v8f acc0 = {}, acc1 = {};
    #pragma unroll
    for (int kc = 0; kc < 8; ++kc) {
      const int abase = ln * ZSTR + kc * 32 + lg * 16;
      v8h a0lo = *(const v8h*)&zl[0][abase];
      v8h a0hi = *(const v8h*)&zl[0][abase + 8];
      v8h a1lo = *(const v8h*)&zl[1][abase];
      v8h a1hi = *(const v8h*)&zl[1][abase + 8];
      const int bbase = (n0 + ln) * ZK + kc * 32 + lg * 16;
      v8h blo = *(const v8h*)&W1T[bbase];
      v8h bhi = *(const v8h*)&W1T[bbase + 8];
      v16h afr0 = combine8(a0lo, a0hi);
      v16h afr1 = combine8(a1lo, a1hi);
      v16h bfr  = combine8(blo, bhi);
      acc0 = __builtin_amdgcn_wmma_f32_16x16x32_f16(false, afr0, false, bfr,
                                                    (short)0, acc0, false, false);
      acc1 = __builtin_amdgcn_wmma_f32_16x16x32_f16(false, afr1, false, bfr,
                                                    (short)0, acc1, false, false);
    }
    const int   ncol = n0 + ln;
    const float be   = b1e[ncol];
    const float sn   = s[ncol];
    #pragma unroll
    for (int v = 0; v < 8; ++v) {
      const int m = v + 8 * lg;          // C layout: lanes16-31 hold M+8
      float a0 = tanh_fast(acc0[v] + be);
      float a1 = tanh_fast(acc1[v] + be);
      divacc[0][v] += (1.0f - a0 * a0) * sn;
      divacc[1][v] += (1.0f - a1 * a1) * sn;
      al[0][m * ASTR + ncol] = (_Float16)a0;
      al[1][m * ASTR + ncol] = (_Float16)a1;
    }
  }

  // divergence: reduce over the 16-lane N-group, then one LDS atomic per (ctx,row)
  #pragma unroll
  for (int c = 0; c < 2; ++c)
    #pragma unroll
    for (int v = 0; v < 8; ++v) {
      float d = divacc[c][v];
      d += __shfl_xor(d, 1, 32);
      d += __shfl_xor(d, 2, 32);
      d += __shfl_xor(d, 4, 32);
      d += __shfl_xor(d, 8, 32);
      if (ln == 0) atomicAdd(&sdiv[c][v + 8 * lg], d);
    }
  __syncthreads();

  // ---- GEMM2: a(16x512) @ W2(512x128); wave owns one 16-col output tile ----
  const int n0b = wave * 16;
  v8f vacc0 = {}, vacc1 = {};
  #pragma unroll
  for (int kc = 0; kc < 16; ++kc) {
    const int abase = ln * ASTR + kc * 32 + lg * 16;
    v8h a0lo = *(const v8h*)&al[0][abase];
    v8h a0hi = *(const v8h*)&al[0][abase + 8];
    v8h a1lo = *(const v8h*)&al[1][abase];
    v8h a1hi = *(const v8h*)&al[1][abase + 8];
    const int bbase = (n0b + ln) * HID + kc * 32 + lg * 16;
    v8h blo = *(const v8h*)&W2T[bbase];
    v8h bhi = *(const v8h*)&W2T[bbase + 8];
    v16h afr0 = combine8(a0lo, a0hi);
    v16h afr1 = combine8(a1lo, a1hi);
    v16h bfr  = combine8(blo, bhi);
    vacc0 = __builtin_amdgcn_wmma_f32_16x16x32_f16(false, afr0, false, bfr,
                                                   (short)0, vacc0, false, false);
    vacc1 = __builtin_amdgcn_wmma_f32_16x16x32_f16(false, afr1, false, bfr,
                                                   (short)0, vacc1, false, false);
  }
  const float gs  = gsp[0];
  const int   nc2 = n0b + ln;
  const float bb  = b2[nc2];
  #pragma unroll
  for (int v = 0; v < 8; ++v) {
    const int m = v + 8 * lg;
    // v = (1-gs)*v_null + gs*v_h ; ctx0=h, ctx1=h_null
    float val = (1.0f - gs) * vacc1[v] + gs * vacc0[v] + bb;
    out[(r0 + m) * (DIM_X + 1) + nc2] = val;
  }
  __syncthreads();
  if (tid < 16) {
    float dv = (1.0f - gs) * sdiv[1][tid] + gs * sdiv[0][tid];
    out[(r0 + tid) * (DIM_X + 1) + DIM_X] = dv;
  }
}

extern "C" void kernel_launch(void* const* d_in, const int* in_sizes, int n_in,
                              void* d_out, int out_size, void* d_ws, size_t ws_size,
                              hipStream_t stream) {
  (void)in_sizes; (void)n_in; (void)out_size; (void)ws_size;
  const float* state = (const float*)d_in[0];
  const float* h     = (const float*)d_in[1];
  const float* hnull = (const float*)d_in[2];
  const float* t     = (const float*)d_in[3];
  const float* gs    = (const float*)d_in[4];
  const float* W1    = (const float*)d_in[5];
  const float* b1    = (const float*)d_in[6];
  const float* W2    = (const float*)d_in[7];
  const float* b2    = (const float*)d_in[8];

  char* ws = (char*)d_ws;
  _Float16* W1T = (_Float16*)(ws + WS_W1T);
  _Float16* W2T = (_Float16*)(ws + WS_W2T);
  float*    b1e = (float*)(ws + WS_B1E);
  float*    s   = (float*)(ws + WS_S);

  hipLaunchKernelGGL(prep_kernel, dim3(512), dim3(128), 0, stream,
                     W1, b1, W2, t, W1T, W2T, b1e, s);
  hipLaunchKernelGGL(fused_cfg_kernel, dim3(BATCH / 16), dim3(256), 0, stream,
                     state, h, hnull, gs, b2, W1T, W2T, b1e, s, (float*)d_out);
}
